// ExactAttention_68917045232093
// MI455X (gfx1250) — compile-verified, matrix operand order checked by reading, not run on hardware
//
#include <hip/hip_runtime.h>

#define NROWS 12288
#define DDIM  128
#define NT16  (NROWS / 16)   // 768 16-row tiles
#define NSLOT (NT16 / 4)     // 192 64-row quads (pass-1 max slots)

typedef __attribute__((ext_vector_type(16))) __bf16 v16bf;
typedef __attribute__((ext_vector_type(8)))  float  v8f;

union FragU { v16bf v; uint4 u[2]; };

// Load a 16-bit A/B fragment: 8 contiguous bf16 at p, 8 contiguous bf16 at p+16
// (CDNA5 16-bit 16x32 layout: VGPR0-3 = k[off..off+7], VGPR4-7 = k[off+16..off+23])
__device__ __forceinline__ v16bf ld_frag(const unsigned short* p) {
  FragU r;
  r.u[0] = *(const uint4*)(p);
  r.u[1] = *(const uint4*)(p + 16);
  return r.v;
}

__device__ __forceinline__ v8f wmma_bf16(v16bf a, v16bf b, v8f c) {
  return __builtin_amdgcn_wmma_f32_16x16x32_bf16(false, a, false, b, (short)0, c, false, false);
}

__device__ __forceinline__ unsigned short f2bf(float f) {
  unsigned u = __float_as_uint(f);
  u += 0x7FFFu + ((u >> 16) & 1u);   // round to nearest even
  return (unsigned short)(u >> 16);
}

// ---------------- pass 0: convert to bf16 (Q, K row-major; V transposed) ----
__global__ void cvt_kernel(const float* __restrict__ Q, const float* __restrict__ K,
                           const float* __restrict__ V,
                           unsigned short* __restrict__ Qb, unsigned short* __restrict__ Kb,
                           unsigned short* __restrict__ Vt) {
  int idx = blockIdx.x * blockDim.x + threadIdx.x;
  if (idx >= NROWS * DDIM) return;
  Qb[idx] = f2bf(Q[idx]);
  Kb[idx] = f2bf(K[idx]);
  int n = idx / DDIM, d = idx % DDIM;
  Vt[d * NROWS + n] = f2bf(V[idx]);
}

__global__ void init_kernel(unsigned* __restrict__ slots) {
  int i = blockIdx.x * blockDim.x + threadIdx.x;
  if (i < NSLOT) slots[i] = 0u;   // minimal order-key
}

// ---------------- pass 1: global max of Q K^T (register-blocked sweep) ------
// One wave computes a 64x128 block: A (4 row-tiles) loaded once, 8 B loads
// amortized over 16 WMMAs per column tile; one reduction + atomic per wave.
__global__ void __launch_bounds__(256)
maxsweep_kernel(const unsigned short* __restrict__ Qb, const unsigned short* __restrict__ Kb,
                unsigned* __restrict__ slots) {
  const int lane = threadIdx.x & 31;
  const int wave = threadIdx.x >> 5;
  const int mrow = lane & 15;
  const int koff = (lane < 16) ? 0 : 8;
  const int rq = blockIdx.y;                 // rows  rq*64  .. rq*64+63
  const int cg = blockIdx.x * 8 + wave;      // cols  cg*128 .. cg*128+127

  const unsigned short* qp = Qb + (size_t)(rq * 64 + mrow) * DDIM + koff;
  const unsigned short* kp = Kb + (size_t)(cg * 128 + mrow) * DDIM + koff;

  v16bf A[4][4];
#pragma unroll
  for (int rt = 0; rt < 4; ++rt)
#pragma unroll
    for (int kk = 0; kk < 4; ++kk)
      A[rt][kk] = ld_frag(qp + rt * 16 * DDIM + kk * 32);

  const v8f zero = {0.f,0.f,0.f,0.f,0.f,0.f,0.f,0.f};
  v8f mx = {-3e38f,-3e38f,-3e38f,-3e38f,-3e38f,-3e38f,-3e38f,-3e38f};

  for (int ct = 0; ct < 8; ++ct) {
    v16bf B[4];
#pragma unroll
    for (int kk = 0; kk < 4; ++kk)
      B[kk] = ld_frag(kp + ct * 16 * DDIM + kk * 32);
    v8f c[4] = {zero, zero, zero, zero};
#pragma unroll
    for (int kk = 0; kk < 4; ++kk)            // 4 independent accumulation chains
#pragma unroll
      for (int rt = 0; rt < 4; ++rt)
        c[rt] = wmma_bf16(A[rt][kk], B[kk], c[rt]);
#pragma unroll
    for (int rt = 0; rt < 4; ++rt)
#pragma unroll
      for (int v = 0; v < 8; ++v)
        mx[v] = fmaxf(mx[v], c[rt][v]);
  }

  float m = mx[0];
#pragma unroll
  for (int v = 1; v < 8; ++v) m = fmaxf(m, mx[v]);
#pragma unroll
  for (int off = 16; off >= 1; off >>= 1) m = fmaxf(m, __shfl_xor(m, off, 32));
  if (lane == 0) {
    unsigned b = __float_as_uint(m);
    unsigned key = (b & 0x80000000u) ? ~b : (b | 0x80000000u);  // order-preserving
    atomicMax(&slots[rq], key);
  }
}

// ---------------- pass 2: fused masked softmax-attention --------------------
#define PST 40   // LDS P-tile row stride in bf16 elements (padded, 16B aligned)

__global__ void __launch_bounds__(32)
attn_kernel(const unsigned short* __restrict__ Qb, const unsigned short* __restrict__ Kb,
            const unsigned short* __restrict__ Vt, const int* __restrict__ seg,
            const unsigned* __restrict__ slots, float* __restrict__ out) {
  __shared__ __align__(16) unsigned short Pl[16 * PST];

  const int lane = threadIdx.x;
  const int r0 = blockIdx.x * 16;
  const int mrow = lane & 15;
  const int koff = (lane < 16) ? 0 : 8;
  const int mbase = (lane < 16) ? 0 : 8;
  const float scale = 0.08838834764831843f;  // 1/sqrt(128)

  // global max from per-quad slots
  float gm = -3.0e38f;
  for (int i = lane; i < NSLOT; i += 32) {
    unsigned k = slots[i];
    float f = (k & 0x80000000u) ? __uint_as_float(k ^ 0x80000000u) : __uint_as_float(~k);
    gm = fmaxf(gm, f);
  }
#pragma unroll
  for (int off = 16; off >= 1; off >>= 1) gm = fmaxf(gm, __shfl_xor(gm, off, 32));

  // per-lane row segments (C/D layout rows)
  int segr[8];
#pragma unroll
  for (int v = 0; v < 8; ++v) segr[v] = seg[r0 + mbase + v];

  // sorted batch_seg => valid columns form one contiguous range; binary search it
  const int seglo = seg[r0], seghi = seg[r0 + 15];
  int lo = 0, hi = NROWS;
  while (lo < hi) { int mid = (lo + hi) >> 1; if (seg[mid] < seglo) lo = mid + 1; else hi = mid; }
  const int cbeg = lo & ~31;
  lo = 0; hi = NROWS;
  while (lo < hi) { int mid = (lo + hi) >> 1; if (seg[mid] <= seghi) lo = mid + 1; else hi = mid; }
  const int cend = lo;

  // Q A-fragments for the whole D=128
  v16bf a[4];
  const unsigned short* qp = Qb + (size_t)(r0 + mrow) * DDIM + koff;
#pragma unroll
  for (int kk = 0; kk < 4; ++kk) a[kk] = ld_frag(qp + kk * 32);

  const v8f zero = {0.f,0.f,0.f,0.f,0.f,0.f,0.f,0.f};
  v8f o[8];
#pragma unroll
  for (int t = 0; t < 8; ++t) o[t] = zero;
  float rs[8] = {0.f,0.f,0.f,0.f,0.f,0.f,0.f,0.f};

  for (int c0 = cbeg; c0 < cend; c0 += 32) {
    const int sc0 = seg[c0 + mrow];
    const int sc1 = seg[c0 + 16 + mrow];
    const unsigned short* kp0 = Kb + (size_t)(c0 + mrow) * DDIM + koff;
    const unsigned short* kp1 = Kb + (size_t)(c0 + 16 + mrow) * DDIM + koff;

    v8f s0 = zero, s1 = zero;
#pragma unroll
    for (int kk = 0; kk < 4; ++kk) s0 = wmma_bf16(a[kk], ld_frag(kp0 + kk * 32), s0);
#pragma unroll
    for (int kk = 0; kk < 4; ++kk) s1 = wmma_bf16(a[kk], ld_frag(kp1 + kk * 32), s1);

    // mask + exp + row-sum, stage P tile (16x32 bf16) in LDS
#pragma unroll
    for (int v = 0; v < 8; ++v) {
      float x0 = (sc0 == segr[v]) ? __expf((s0[v] - gm) * scale) : 0.0f;
      float x1 = (sc1 == segr[v]) ? __expf((s1[v] - gm) * scale) : 0.0f;
      rs[v] += x0 + x1;
      int m = mbase + v;
      Pl[m * PST + mrow]      = f2bf(x0);
      Pl[m * PST + mrow + 16] = f2bf(x1);
    }
    __syncthreads();

    // reload P in A-fragment layout (two b128 DS loads)
    FragU pf;
    const unsigned short* pp = &Pl[mrow * PST + koff];
    pf.u[0] = *(const uint4*)(pp);
    pf.u[1] = *(const uint4*)(pp + 16);
    v16bf p = pf.v;

    // O += P * V  (V transposed => contiguous B-fragment loads)
#pragma unroll
    for (int t = 0; t < 8; ++t) {
      const unsigned short* vp = Vt + (size_t)(t * 16 + mrow) * NROWS + c0 + koff;
      o[t] = wmma_bf16(p, ld_frag(vp), o[t]);
    }
    __syncthreads();
  }

  // reduce row sums across the 16 lanes sharing each row (masks <=8 stay in half)
#pragma unroll
  for (int v = 0; v < 8; ++v) {
    float r = rs[v];
#pragma unroll
    for (int off = 8; off >= 1; off >>= 1) r += __shfl_xor(r, off, 32);
    rs[v] = 1.0f / (r + 1e-8f);
  }

  // write normalized output
#pragma unroll
  for (int t = 0; t < 8; ++t) {
#pragma unroll
    for (int v = 0; v < 8; ++v) {
      int m = mbase + v;
      out[(size_t)(r0 + m) * DDIM + t * 16 + mrow] = o[t][v] * rs[v];
    }
  }
}

// ---------------------------------------------------------------------------
extern "C" void kernel_launch(void* const* d_in, const int* in_sizes, int n_in,
                              void* d_out, int out_size, void* d_ws, size_t ws_size,
                              hipStream_t stream) {
  const float* Q = (const float*)d_in[0];
  const float* K = (const float*)d_in[1];
  const float* V = (const float*)d_in[2];
  const int* seg = (const int*)d_in[4];
  float* out = (float*)d_out;

  unsigned short* Qb = (unsigned short*)d_ws;
  unsigned short* Kb = Qb + (size_t)NROWS * DDIM;
  unsigned short* Vt = Kb + (size_t)NROWS * DDIM;
  unsigned* slots = (unsigned*)(Vt + (size_t)NROWS * DDIM);

  cvt_kernel<<<dim3((NROWS * DDIM + 255) / 256), 256, 0, stream>>>(Q, K, V, Qb, Kb, Vt);
  init_kernel<<<dim3(1), 256, 0, stream>>>(slots);
  // 64x128 per wave: grid = (768/8/8 col-groups-of-8-waves = 12, 768/4 = 192)
  maxsweep_kernel<<<dim3(12, NSLOT), 256, 0, stream>>>(Qb, Kb, slots);
  attn_kernel<<<dim3(NT16), 32, 0, stream>>>(Qb, Kb, Vt, seg, slots, out);
}
